// HeFF_81088982548725
// MI455X (gfx1250) — compile-verified, matrix operand order checked by reading, not run on hardware
//
#include <hip/hip_runtime.h>
#include <hip/hip_bf16.h>
#include <math.h>

typedef __bf16 bf16_t;
typedef __attribute__((ext_vector_type(16))) __bf16 v16bf;
typedef __attribute__((ext_vector_type(8)))  __bf16 v8bf;
typedef __attribute__((ext_vector_type(8)))  float  v8f;

#define NNODES 50000
#define NEDGES 600000
#define DIM    128
#define NTYPE  4
#define NREL   4
#define NHEAD  8
#define DKH    16
#define MAXLEN 240

// ---------- helpers: order-preserving float<->uint key for atomic max ----------
__device__ __forceinline__ unsigned fkey(float f) {
  unsigned u = __float_as_uint(f);
  return (u & 0x80000000u) ? ~u : (u | 0x80000000u);
}
__device__ __forceinline__ float fkey_inv(unsigned k) {
  unsigned u = (k & 0x80000000u) ? (k & 0x7fffffffu) : ~k;
  return __uint_as_float(u);
}

// ---------- pack weights: f32 [T,in,out] -> bf16 transposed [T,out,in] ----------
__global__ void hgt_pack_weights(const float* __restrict__ Wk, const float* __restrict__ Wq,
                                 const float* __restrict__ Wv, const float* __restrict__ Wa,
                                 bf16_t* __restrict__ wT) {
  int idx = blockIdx.x * 256 + threadIdx.x;
  if (idx >= 4 * NTYPE * DIM * DIM) return;
  int w   = idx >> 16;          // T*D*D = 65536
  int rem = idx & 65535;
  int t   = rem >> 14;          // D*D = 16384
  int io  = rem & 16383;
  int o   = io >> 7, i = io & 127;
  const float* S = (w == 0) ? Wk : (w == 1) ? Wq : (w == 2) ? Wv : Wa;
  wT[idx] = (bf16_t)S[t * 16384 + i * 128 + o];
}

__global__ void hgt_x_to_bf16(const float* __restrict__ x, bf16_t* __restrict__ xb) {
  int i = blockIdx.x * 256 + threadIdx.x;
  if (i < NNODES * DIM) xb[i] = (bf16_t)x[i];
}

// ---------- RTE: sinusoid table @ rte_W + rte_b -> rte_out [240,128] ----------
__global__ void hgt_rte_table(const float* __restrict__ rteW, const float* __restrict__ rteb,
                              float* __restrict__ rte_out) {
  __shared__ float emb[DIM];
  int m = blockIdx.x, d = threadIdx.x;
  int j = d & ~1;
  float freq = __expf((float)j * (-9.210340371976184f / 128.0f)); // -ln(10000)/D
  float ang  = (float)m * freq;
  emb[d] = ((d & 1) ? __cosf(ang) : __sinf(ang)) * 0.08838834764831845f; // 1/sqrt(128)
  __syncthreads();
  float acc = rteb[d];
  for (int i = 0; i < DIM; ++i) acc += emb[i] * rteW[i * DIM + d];
  rte_out[m * DIM + d] = acc;
}

// ---------- k_rte/v_rte[t,m,:] = rte_out[m,:] @ W{k,v}[t] (no bias) ----------
__global__ void hgt_rte_proj(const float* __restrict__ rte_out, const float* __restrict__ Wk,
                             const float* __restrict__ Wv, float* __restrict__ k_rte,
                             float* __restrict__ v_rte) {
  __shared__ float row[DIM];
  int t = blockIdx.x / MAXLEN, m = blockIdx.x % MAXLEN, o = threadIdx.x;
  row[o] = rte_out[m * DIM + o];
  __syncthreads();
  float ak = 0.f, av = 0.f;
  const float* wk = Wk + (size_t)t * DIM * DIM;
  const float* wv = Wv + (size_t)t * DIM * DIM;
  for (int i = 0; i < DIM; ++i) { float r = row[i]; ak += r * wk[i * DIM + o]; av += r * wv[i * DIM + o]; }
  k_rte[((size_t)t * MAXLEN + m) * DIM + o] = ak;
  v_rte[((size_t)t * MAXLEN + m) * DIM + o] = av;
}

// ---------- per-type GEMM via WMMA: out[n,:] = x[n,:] @ W[type[n]] + b[type[n]] ----------
// One wave per (16-row tile, 16-col tile). C = sum_t mask_t(A) x W_t  (bf16 WMMA, f32 accum)
__global__ void __launch_bounds__(32)
hgt_typed_gemm(const bf16_t* __restrict__ X, const bf16_t* __restrict__ Wt,
               const float* __restrict__ bias, const int* __restrict__ types,
               float* __restrict__ out) {
  const int tile = blockIdx.x, ntile = blockIdx.y;
  const int lane = threadIdx.x;
  const int half = lane >> 4, r = lane & 15;

  union F { v16bf v; v8bf h[2]; };
  // A fragments per ISA 16-bit 16x32 layout: lane r (both halves) holds row r,
  // half 0 -> K {k0..k0+7, k0+16..k0+23}, half 1 -> +8.
  const bf16_t* arow = X + (size_t)(tile * 16 + r) * DIM;
  F a[4];
  #pragma unroll
  for (int ks = 0; ks < 4; ++ks) {
    int k0 = ks * 32 + half * 8;
    a[ks].h[0] = *reinterpret_cast<const v8bf*>(arow + k0);
    a[ks].h[1] = *reinterpret_cast<const v8bf*>(arow + k0 + 16);
  }
  const int rtype = types[tile * 16 + r];
  v8f c = {};
  v16bf zero = {};
  #pragma unroll
  for (int t = 0; t < NTYPE; ++t) {
    if (!__any(rtype == t)) continue;   // wave-uniform skip, EXEC stays all-ones
    const bf16_t* brow = Wt + (size_t)(t * DIM + ntile * 16 + r) * DIM; // transposed [out][in]
    #pragma unroll
    for (int ks = 0; ks < 4; ++ks) {
      F b;
      int k0 = ks * 32 + half * 8;
      b.h[0] = *reinterpret_cast<const v8bf*>(brow + k0);
      b.h[1] = *reinterpret_cast<const v8bf*>(brow + k0 + 16);
      v16bf am = (rtype == t) ? a[ks].v : zero;
      c = __builtin_amdgcn_wmma_f32_16x16x32_bf16(false, am, false, b.v,
                                                  (short)0, c, false, false);
    }
  }
  // C layout: VGPR v -> row (v + 8*half), lane%16 -> col
  const int col = ntile * 16 + r;
  #pragma unroll
  for (int v = 0; v < 8; ++v) {
    int row = tile * 16 + half * 8 + v;
    int ty  = types[row];
    out[(size_t)row * DIM + col] = c[v] + bias[ty * DIM + col];
  }
}

// ---------- edge pass 1: logits + segment max ----------
__global__ void hgt_edge_logits(const int* __restrict__ ei, const int* __restrict__ etype,
                                const int* __restrict__ etime, const int* __restrict__ ntype,
                                const float* __restrict__ q_node, const float* __restrict__ k_node,
                                const float* __restrict__ k_rte, const float* __restrict__ rel_att,
                                const float* __restrict__ rel_pri,
                                float* __restrict__ logits, unsigned* __restrict__ segmax) {
  __shared__ float satt[NREL * NHEAD * DKH * DKH];   // 32 KB
  int tid = threadIdx.x;
  for (int i = tid; i < NREL * NHEAD * DKH * DKH; i += 256) satt[i] = rel_att[i];
  __syncthreads();
  int gid = blockIdx.x * 256 + tid;              // E*H divides 256 exactly
  int e = gid >> 3, h = gid & 7;
  int src = ei[e], tgt = ei[NEDGES + e];
  int rr = etype[e], tm = etime[e], st = ntype[src];
  const float* qp = q_node + (size_t)tgt * DIM + h * DKH;
  const float* kp = k_node + (size_t)src * DIM + h * DKH;
  const float* rp = k_rte + ((size_t)st * MAXLEN + tm) * DIM + h * DKH;
  float kk[DKH];
  #pragma unroll
  for (int k = 0; k < DKH; ++k) kk[k] = kp[k] + rp[k];
  const float* A = satt + (rr * NHEAD + h) * DKH * DKH;
  float acc = 0.f;
  #pragma unroll
  for (int d = 0; d < DKH; ++d) {
    float t2 = 0.f;
    #pragma unroll
    for (int k = 0; k < DKH; ++k) t2 += kk[k] * A[k * DKH + d];
    acc += qp[d] * t2;
  }
  acc *= rel_pri[rr * NHEAD + h] * 0.25f;        // 1/sqrt(DK)
  logits[(size_t)e * NHEAD + h] = acc;
  atomicMax(segmax + (size_t)tgt * NHEAD + h, fkey(acc));
}

// ---------- edge pass 2: ex = exp(logit - max); den += ex ----------
__global__ void hgt_edge_expsum(const int* __restrict__ ei, float* __restrict__ logits,
                                const unsigned* __restrict__ segmax, float* __restrict__ den) {
  int gid = blockIdx.x * 256 + threadIdx.x;
  int e = gid >> 3, h = gid & 7;
  int tgt = ei[NEDGES + e];
  float m  = fkey_inv(segmax[(size_t)tgt * NHEAD + h]);
  float ex = __expf(logits[(size_t)e * NHEAD + h] - m);
  logits[(size_t)e * NHEAD + h] = ex;
  atomicAdd(den + (size_t)tgt * NHEAD + h, ex);
}

// ---------- edge pass 3: msg = v_e @ rel_msg[r]; aggr[tgt] += att * msg ----------
__global__ void hgt_edge_aggregate(const int* __restrict__ ei, const int* __restrict__ etype,
                                   const int* __restrict__ etime, const int* __restrict__ ntype,
                                   const float* __restrict__ v_node, const float* __restrict__ v_rte,
                                   const float* __restrict__ rel_msg, const float* __restrict__ logits,
                                   const float* __restrict__ den, float* __restrict__ aggr) {
  __shared__ float smsg[NREL * NHEAD * DKH * DKH];
  int tid = threadIdx.x;
  for (int i = tid; i < NREL * NHEAD * DKH * DKH; i += 256) smsg[i] = rel_msg[i];
  __syncthreads();
  int gid = blockIdx.x * 256 + tid;
  int e = gid >> 3, h = gid & 7;
  int src = ei[e], tgt = ei[NEDGES + e];
  int rr = etype[e], tm = etime[e], st = ntype[src];
  float att = logits[(size_t)e * NHEAD + h] / den[(size_t)tgt * NHEAD + h];
  const float* vp = v_node + (size_t)src * DIM + h * DKH;
  const float* rp = v_rte + ((size_t)st * MAXLEN + tm) * DIM + h * DKH;
  float vv[DKH];
  #pragma unroll
  for (int k = 0; k < DKH; ++k) vv[k] = vp[k] + rp[k];
  const float* M = smsg + (rr * NHEAD + h) * DKH * DKH;
  float* op = aggr + (size_t)tgt * DIM + h * DKH;
  #pragma unroll
  for (int d = 0; d < DKH; ++d) {
    float ms = 0.f;
    #pragma unroll
    for (int k = 0; k < DKH; ++k) ms += vv[k] * M[k * DKH + d];
    atomicAdd(op + d, att * ms);
  }
}

// ---------- exact GELU(aggr) -> bf16 (input to update GEMM) ----------
__global__ void hgt_gelu_bf16(const float* __restrict__ aggr, bf16_t* __restrict__ ab) {
  int i = blockIdx.x * 256 + threadIdx.x;
  if (i < NNODES * DIM) {
    float a = aggr[i];
    float g = 0.5f * a * (1.f + erff(a * 0.7071067811865476f));
    ab[i] = (bf16_t)g;
  }
}

// ---------- gated skip + per-type LayerNorm; one wave32 per node row ----------
__global__ void hgt_skip_layernorm(const float* __restrict__ trans, const float* __restrict__ x,
                                   const int* __restrict__ ntype, const float* __restrict__ skip,
                                   const float* __restrict__ gamma, const float* __restrict__ beta,
                                   float* __restrict__ out) {
  int tid  = threadIdx.x;
  int n    = blockIdx.x * 8 + (tid >> 5);
  int lane = tid & 31;
  int t = ntype[n];
  float alpha = 1.f / (1.f + __expf(-skip[t]));
  float h[4]; float s = 0.f, s2 = 0.f;
  #pragma unroll
  for (int j = 0; j < 4; ++j) {
    int c = lane + 32 * j;
    float hv = trans[(size_t)n * DIM + c] * alpha + x[(size_t)n * DIM + c] * (1.f - alpha);
    h[j] = hv; s += hv; s2 += hv * hv;
  }
  #pragma unroll
  for (int off = 16; off > 0; off >>= 1) {   // wave32 tree reduce
    s  += __shfl_xor(s,  off);
    s2 += __shfl_xor(s2, off);
  }
  float mu  = s  * (1.f / 128.f);
  float var = s2 * (1.f / 128.f) - mu * mu;
  float inv = rsqrtf(var + 1e-5f);
  #pragma unroll
  for (int j = 0; j < 4; ++j) {
    int c = lane + 32 * j;
    out[(size_t)n * DIM + c] = (h[j] - mu) * inv * gamma[t * DIM + c] + beta[t * DIM + c];
  }
}

extern "C" void kernel_launch(void* const* d_in, const int* in_sizes, int n_in,
                              void* d_out, int out_size, void* d_ws, size_t ws_size,
                              hipStream_t stream) {
  (void)in_sizes; (void)n_in; (void)out_size; (void)ws_size;
  const float* node_inp  = (const float*)d_in[0];
  const int*   node_type = (const int*)d_in[1];
  const int*   edge_index= (const int*)d_in[2];
  const int*   edge_type = (const int*)d_in[3];
  const int*   edge_time = (const int*)d_in[4];
  const float* Wk = (const float*)d_in[5];
  const float* bk = (const float*)d_in[6];
  const float* Wq = (const float*)d_in[7];
  const float* bq = (const float*)d_in[8];
  const float* Wv = (const float*)d_in[9];
  const float* bv = (const float*)d_in[10];
  const float* Wa = (const float*)d_in[11];
  const float* ba = (const float*)d_in[12];
  const float* gamma = (const float*)d_in[13];
  const float* beta  = (const float*)d_in[14];
  const float* rel_pri = (const float*)d_in[15];
  const float* rel_att = (const float*)d_in[16];
  const float* rel_msg = (const float*)d_in[17];
  const float* skip = (const float*)d_in[18];
  const float* rteW = (const float*)d_in[19];
  const float* rteb = (const float*)d_in[20];
  float* out = (float*)d_out;

  char* ws = (char*)d_ws;
  size_t off = 0;
  auto take = [&](size_t bytes) -> char* {
    char* p = ws + off; off = (off + bytes + 255) & ~(size_t)255; return p;
  };
  bf16_t*  xb      = (bf16_t*)take((size_t)NNODES * DIM * 2);          // also reused for gelu bf16
  bf16_t*  wT      = (bf16_t*)take((size_t)4 * NTYPE * DIM * DIM * 2); // [Wk|Wq|Wv|Wa] transposed
  float*   k_node  = (float*)take((size_t)NNODES * DIM * 4);           // reused as 'trans'
  float*   q_node  = (float*)take((size_t)NNODES * DIM * 4);
  float*   v_node  = (float*)take((size_t)NNODES * DIM * 4);
  float*   rte_out = (float*)take((size_t)MAXLEN * DIM * 4);
  float*   k_rte   = (float*)take((size_t)NTYPE * MAXLEN * DIM * 4);
  float*   v_rte   = (float*)take((size_t)NTYPE * MAXLEN * DIM * 4);
  float*   logits  = (float*)take((size_t)NEDGES * NHEAD * 4);
  unsigned* segmax = (unsigned*)take((size_t)NNODES * NHEAD * 4);
  float*   den     = (float*)take((size_t)NNODES * NHEAD * 4);
  float*   aggr    = (float*)take((size_t)NNODES * DIM * 4);

  // key 0 == encoded -NaN < any finite key, valid "-inf" init for atomic max
  hipMemsetAsync(segmax, 0, (size_t)NNODES * NHEAD * 4, stream);
  hipMemsetAsync(den,    0, (size_t)NNODES * NHEAD * 4, stream);
  hipMemsetAsync(aggr,   0, (size_t)NNODES * DIM * 4,  stream);

  hgt_pack_weights<<<(4 * NTYPE * DIM * DIM + 255) / 256, 256, 0, stream>>>(Wk, Wq, Wv, Wa, wT);
  hgt_x_to_bf16<<<(NNODES * DIM + 255) / 256, 256, 0, stream>>>(node_inp, xb);
  hgt_rte_table<<<MAXLEN, DIM, 0, stream>>>(rteW, rteb, rte_out);
  hgt_rte_proj<<<NTYPE * MAXLEN, DIM, 0, stream>>>(rte_out, Wk, Wv, k_rte, v_rte);

  dim3 ggrid(NNODES / 16, DIM / 16);   // 3125 x 8, one wave32 per block
  hgt_typed_gemm<<<ggrid, 32, 0, stream>>>(xb, wT + (size_t)0 * NTYPE * DIM * DIM, bk, node_type, k_node);
  hgt_typed_gemm<<<ggrid, 32, 0, stream>>>(xb, wT + (size_t)1 * NTYPE * DIM * DIM, bq, node_type, q_node);
  hgt_typed_gemm<<<ggrid, 32, 0, stream>>>(xb, wT + (size_t)2 * NTYPE * DIM * DIM, bv, node_type, v_node);

  int eblocks = (NEDGES * NHEAD) / 256;  // exact
  hgt_edge_logits<<<eblocks, 256, 0, stream>>>(edge_index, edge_type, edge_time, node_type,
                                               q_node, k_node, k_rte, rel_att, rel_pri,
                                               logits, segmax);
  hgt_edge_expsum<<<eblocks, 256, 0, stream>>>(edge_index, logits, segmax, den);
  hgt_edge_aggregate<<<eblocks, 256, 0, stream>>>(edge_index, edge_type, edge_time, node_type,
                                                  v_node, v_rte, rel_msg, logits, den, aggr);

  hgt_gelu_bf16<<<(NNODES * DIM + 255) / 256, 256, 0, stream>>>(aggr, xb);
  hgt_typed_gemm<<<ggrid, 32, 0, stream>>>(xb, wT + (size_t)3 * NTYPE * DIM * DIM, ba, node_type,
                                           k_node /* trans (k_node is dead) */);
  hgt_skip_layernorm<<<NNODES / 8, 256, 0, stream>>>(k_node, node_inp, node_type, skip,
                                                     gamma, beta, out);
}
